// LSTMClassifier_70961449665085
// MI455X (gfx1250) — matrix operand order repaired and checked by reading, not compiled, not run on hardware
//
#include <hip/hip_runtime.h>
#include <hip/hip_bf16.h>

typedef __attribute__((ext_vector_type(16))) __bf16 v16bf;
typedef __attribute__((ext_vector_type(8)))  __bf16 v8bf;
typedef __attribute__((ext_vector_type(8)))  float  v8f;

#define T_STEPS 512
#define BATCH   256
#define DIN     256
#define HDIM    128
#define G4H     512   // 4*H

// workspace layout (bytes)
#define WIH_OFF  0u           // 512*256 bf16 = 256 KB
#define WHH_OFF  262144u      // 512*128 bf16 = 128 KB
#define BIAS_OFF 393216u      // 512 f32
#define XG_OFF   524288u      // T*B*4H f32 = 256 MB

// native HW convert (backend selects v_cvt*bf16* and packs pairs)
__device__ __forceinline__ unsigned short f32_to_bf16(float f) {
  __bf16 b = (__bf16)f;
  return __builtin_bit_cast(unsigned short, b);
}
__device__ __forceinline__ float sigm(float x) { return 1.0f / (1.0f + __expf(-x)); }
__device__ __forceinline__ float tanh_fast(float x) {
  x = fminf(fmaxf(x, -15.0f), 15.0f);
  float e = __expf(2.0f * x);
  return (e - 1.0f) / (e + 1.0f);
}

union Frag { v16bf v; v8bf h[2]; };

// A fragment: 16(M) x 32(K) bf16, row-major source with pitch (elements).
// lane<16: K[0:8)+K[16:24); lane>=16: K[8:16)+K[24:32)  (ISA 7.12.2)
__device__ __forceinline__ v16bf load_fragA(const unsigned short* p, int pitch,
                                            int rowBase, int k0, int lane) {
  int ln = lane & 15, lh = lane >> 4;
  const unsigned short* r = p + (size_t)(rowBase + ln) * pitch;
  Frag f;
  f.h[0] = *(const v8bf*)(r + k0 + lh * 8);
  f.h[1] = *(const v8bf*)(r + k0 + 16 + lh * 8);
  return f.v;
}
// B fragment: 32(K) x 16(N); source is B^T row-major ([N][K], pitch in elements).
// lane<16: K[0:8)+K[16:24) of column N=lane; lane>=16: K[8:16)+K[24:32) of column N=lane-16.
__device__ __forceinline__ v16bf load_fragB(const unsigned short* p, int pitch,
                                            int rowBase, int k0, int lane) {
  int ln = lane & 15, lh = lane >> 4;
  const unsigned short* r = p + (size_t)(rowBase + ln) * pitch + k0 + lh * 8;
  Frag f;
  f.h[0] = *(const v8bf*)(r);
  f.h[1] = *(const v8bf*)(r + 16);
  return f.v;
}

// ---------------- Kernel 1: convert weights to bf16, fuse biases ----------------
__global__ void prep_kernel(const float* __restrict__ Wih, const float* __restrict__ Whh,
                            const float* __restrict__ bih, const float* __restrict__ bhh,
                            unsigned short* __restrict__ wih_bf,
                            unsigned short* __restrict__ whh_bf,
                            float* __restrict__ bias) {
  int i = blockIdx.x * blockDim.x + threadIdx.x;
  if (i < G4H * DIN) wih_bf[i] = f32_to_bf16(Wih[i]);
  if (i < G4H * HDIM) whh_bf[i] = f32_to_bf16(Whh[i]);
  if (i < G4H) bias[i] = bih[i] + bhh[i];
}

// ---------------- Kernel 2: x_gates = x @ W_ih^T + bias  (bf16 WMMA) ----------------
// Block tile 128(M) x 64(N); K staged through double-buffered LDS in chunks of 64,
// with register staging so next stage's HBM loads overlap this stage's WMMAs.
#define K2P 72   // padded LDS pitch (elements): 144B rows, conflict-free frags
__global__ void __launch_bounds__(256)
xgates_gemm(const float* __restrict__ x, const unsigned short* __restrict__ wih_bf,
            const float* __restrict__ bias, float* __restrict__ xg) {
  __shared__ unsigned short lA[2][128 * K2P];   // 2 x 18432 B
  __shared__ unsigned short lB[2][64 * K2P];    // 2 x  9216 B
  const int tid = threadIdx.x;
  const int lane = tid & 31, w = tid >> 5;
  const int wm = w & 3, wn = w >> 2;            // 4(M) x 2(N) wave grid, 32x32 per wave
  const int m0 = blockIdx.x * 128;
  const int n0 = blockIdx.y * 64;

  float4 ra[8];   // staged A (f32)
  v8bf   rb[2];   // staged B (bf16)

  auto load_stage = [&](int ks) {
    const int kbase = ks * 64;
#pragma unroll
    for (int i = 0; i < 8; ++i) {               // 128 rows x 16 float4
      int q = tid + i * 256;
      int row = q >> 4, cv = q & 15;
      ra[i] = *(const float4*)(x + (size_t)(m0 + row) * DIN + kbase + cv * 4);
    }
#pragma unroll
    for (int i = 0; i < 2; ++i) {               // 64 rows x 8 groups of 8 bf16
      int q = tid + i * 256;
      int row = q >> 3, g = q & 7;
      rb[i] = *(const v8bf*)(wih_bf + (size_t)(n0 + row) * DIN + kbase + g * 8);
    }
  };
  auto store_stage = [&](int buf) {
#pragma unroll
    for (int i = 0; i < 8; ++i) {
      int q = tid + i * 256;
      int row = q >> 4, cv = q & 15;
      unsigned short* d = &lA[buf][row * K2P + cv * 4];
      d[0] = f32_to_bf16(ra[i].x); d[1] = f32_to_bf16(ra[i].y);
      d[2] = f32_to_bf16(ra[i].z); d[3] = f32_to_bf16(ra[i].w);
    }
#pragma unroll
    for (int i = 0; i < 2; ++i) {
      int q = tid + i * 256;
      int row = q >> 3, g = q & 7;
      *(v8bf*)&lB[buf][row * K2P + g * 8] = rb[i];
    }
  };

  v8f acc[2][2];
#pragma unroll
  for (int i = 0; i < 2; ++i)
#pragma unroll
    for (int j = 0; j < 2; ++j)
#pragma unroll
      for (int v = 0; v < 8; ++v) acc[i][j][v] = 0.0f;

  load_stage(0);
#pragma unroll
  for (int ks = 0; ks < 4; ++ks) {
    const int buf = ks & 1;
    store_stage(buf);
    __syncthreads();                 // stores visible; prior reads of buf done
    if (ks < 3) load_stage(ks + 1);  // HBM latency hides behind this stage's WMMAs
#pragma unroll
    for (int kk = 0; kk < 2; ++kk) {
      int k0 = kk * 32;
      v16bf a0 = load_fragA(lA[buf], K2P, 32 * wm,      k0, lane);
      v16bf a1 = load_fragA(lA[buf], K2P, 32 * wm + 16, k0, lane);
      v16bf b0 = load_fragB(lB[buf], K2P, 32 * wn,      k0, lane);
      v16bf b1 = load_fragB(lB[buf], K2P, 32 * wn + 16, k0, lane);
      acc[0][0] = __builtin_amdgcn_wmma_f32_16x16x32_bf16(false, a0, false, b0, (short)0, acc[0][0], false, false);
      acc[0][1] = __builtin_amdgcn_wmma_f32_16x16x32_bf16(false, a0, false, b1, (short)0, acc[0][1], false, false);
      acc[1][0] = __builtin_amdgcn_wmma_f32_16x16x32_bf16(false, a1, false, b0, (short)0, acc[1][0], false, false);
      acc[1][1] = __builtin_amdgcn_wmma_f32_16x16x32_bf16(false, a1, false, b1, (short)0, acc[1][1], false, false);
    }
  }
  // epilogue: add bias, store f32 (D layout: lane%16 -> N, vgpr+8*(lane/16) -> M)
  const int ln = lane & 15, lh = lane >> 4;
#pragma unroll
  for (int mi = 0; mi < 2; ++mi)
#pragma unroll
    for (int ni = 0; ni < 2; ++ni) {
      int ncol = n0 + 32 * wn + 16 * ni + ln;
      float b = bias[ncol];
      int mrow = m0 + 32 * wm + 16 * mi + 8 * lh;
#pragma unroll
      for (int v = 0; v < 8; ++v)
        xg[(size_t)(mrow + v) * G4H + ncol] = acc[mi][ni][v] + b;
    }
}

// ---------------- Kernel 3: persistent recurrence, 16 WGs x 16 batch rows ----------------
// W_hh lives in registers (16 frags = 128 VGPRs), h in LDS (bf16), c in regs.
// x_gates stream is software-pipelined one step ahead + L2-prefetched two ahead.
#define HP 136   // h LDS pitch (elements): 272B rows, conflict-free A-frag reads
__global__ void __launch_bounds__(256)
lstm_recur(const float* __restrict__ xg, const unsigned short* __restrict__ whh_bf,
           const float* __restrict__ h0, const float* __restrict__ c0,
           const float* __restrict__ fcw, const float* __restrict__ fcb,
           float* __restrict__ out) {
  __shared__ unsigned short hbuf[16 * HP];   // h state, bf16
  __shared__ float hfin[16 * HDIM];          // final h, f32
  const int tid = threadIdx.x;
  const int lane = tid & 31, w = tid >> 5;   // wave w owns hidden cols [16w, 16w+16)
  const int ln = lane & 15, lh = lane >> 4;
  const int b0 = blockIdx.x * 16;

  // W_hh fragments resident in registers: 4 gates x 4 K-steps
  v16bf frB[4][4];
#pragma unroll
  for (int g = 0; g < 4; ++g)
#pragma unroll
    for (int kk = 0; kk < 4; ++kk)
      frB[g][kk] = load_fragB(whh_bf, HDIM, g * HDIM + 16 * w, kk * 32, lane);

  // init h (LDS, bf16) and c (regs, accumulator layout)
  float c[8];
#pragma unroll
  for (int v = 0; v < 8; ++v) {
    int m = 8 * lh + v, j = 16 * w + ln;
    c[v] = c0[(size_t)(b0 + m) * HDIM + j];
    hbuf[m * HP + j] = f32_to_bf16(h0[(size_t)(b0 + m) * HDIM + j]);
  }

  // pipeline: preload x_gates for t = 0
  float xn[4][8];
  {
    const float* xgt = xg + (size_t)b0 * G4H;
#pragma unroll
    for (int g = 0; g < 4; ++g) {
      int ncol = g * HDIM + 16 * w + ln;
#pragma unroll
      for (int v = 0; v < 8; ++v)
        xn[g][v] = xgt[(size_t)(8 * lh + v) * G4H + ncol];
    }
  }
  __syncthreads();

  for (int t = 0; t < T_STEPS; ++t) {
    // seed accumulators from the pipelined registers
    v8f acc[4];
#pragma unroll
    for (int g = 0; g < 4; ++g)
#pragma unroll
      for (int v = 0; v < 8; ++v) acc[g][v] = xn[g][v];

    // issue t+1 loads now: latency hides behind WMMAs + activations + barriers
    if (t + 1 < T_STEPS) {
      const float* xgt = xg + (size_t)((t + 1) * BATCH + b0) * G4H;
#pragma unroll
      for (int g = 0; g < 4; ++g) {
        int ncol = g * HDIM + 16 * w + ln;
#pragma unroll
        for (int v = 0; v < 8; ++v)
          xn[g][v] = xgt[(size_t)(8 * lh + v) * G4H + ncol];
      }
    }
    // stage t+2's 32KB block HBM -> L2/L0 (one 128B line per thread)
    if (t + 2 < T_STEPS) {
      const char* pf = (const char*)(xg + (size_t)((t + 2) * BATCH + b0) * G4H) + tid * 128;
      __builtin_prefetch(pf, 0, 1);
    }

    // gates += h @ W_hh^T
#pragma unroll
    for (int kk = 0; kk < 4; ++kk) {
      v16bf a = load_fragA(hbuf, HP, 0, kk * 32, lane);
      acc[0] = __builtin_amdgcn_wmma_f32_16x16x32_bf16(false, a, false, frB[0][kk], (short)0, acc[0], false, false);
      acc[1] = __builtin_amdgcn_wmma_f32_16x16x32_bf16(false, a, false, frB[1][kk], (short)0, acc[1], false, false);
      acc[2] = __builtin_amdgcn_wmma_f32_16x16x32_bf16(false, a, false, frB[2][kk], (short)0, acc[2], false, false);
      acc[3] = __builtin_amdgcn_wmma_f32_16x16x32_bf16(false, a, false, frB[3][kk], (short)0, acc[3], false, false);
    }
    __syncthreads();   // everyone done reading h before overwrite
#pragma unroll
    for (int v = 0; v < 8; ++v) {
      float ig = sigm(acc[0][v]);
      float fg = sigm(acc[1][v]);
      float gg = tanh_fast(acc[2][v]);
      float og = sigm(acc[3][v]);
      c[v] = fg * c[v] + ig * gg;
      float hv = og * tanh_fast(c[v]);
      int m = 8 * lh + v, j = 16 * w + ln;
      hbuf[m * HP + j] = f32_to_bf16(hv);
      if (t == T_STEPS - 1) hfin[m * HDIM + j] = hv;
    }
    __syncthreads();   // new h visible before next step's reads
  }

  // FC head: out[b] = h[b,:] . fc_w + fc_b
  if (tid < 16) {
    float s = fcb[0];
#pragma unroll 8
    for (int j = 0; j < HDIM; ++j) s += hfin[tid * HDIM + j] * fcw[j];
    out[b0 + tid] = s;
  }
}

extern "C" void kernel_launch(void* const* d_in, const int* in_sizes, int n_in,
                              void* d_out, int out_size, void* d_ws, size_t ws_size,
                              hipStream_t stream) {
  const float* x   = (const float*)d_in[0];
  const float* h0  = (const float*)d_in[1];
  const float* c0  = (const float*)d_in[2];
  const float* Wih = (const float*)d_in[3];
  const float* Whh = (const float*)d_in[4];
  const float* bih = (const float*)d_in[5];
  const float* bhh = (const float*)d_in[6];
  const float* fcw = (const float*)d_in[7];
  const float* fcb = (const float*)d_in[8];
  float* out = (float*)d_out;

  char* ws = (char*)d_ws;
  unsigned short* wih_bf = (unsigned short*)(ws + WIH_OFF);
  unsigned short* whh_bf = (unsigned short*)(ws + WHH_OFF);
  float* bias = (float*)(ws + BIAS_OFF);
  float* xg   = (float*)(ws + XG_OFF);

  prep_kernel<<<512, 256, 0, stream>>>(Wih, Whh, bih, bhh, wih_bf, whh_bf, bias);
  dim3 g2((T_STEPS * BATCH) / 128, G4H / 64);   // 1024 x 8
  xgates_gemm<<<g2, 256, 0, stream>>>(x, wih_bf, bias, xg);
  lstm_recur<<<BATCH / 16, 256, 0, stream>>>(xg, whh_bf, h0, c0, fcw, fcb, out);
}
